// GCN_31980326486189
// MI455X (gfx1250) — matrix operand order repaired and checked by reading, not compiled, not run on hardware
//
#include <hip/hip_runtime.h>
#include <math.h>

typedef __attribute__((ext_vector_type(16))) _Float16 v16h;
typedef __attribute__((ext_vector_type(8)))  float    v8f;

#define BN_ROWS 512

// ---------------- utility kernels ----------------

__global__ __launch_bounds__(256) void k_fill(float* __restrict__ p, float v, int n) {
  int i = blockIdx.x * blockDim.x + threadIdx.x;
  if (i < n) p[i] = v;
}

__global__ __launch_bounds__(256) void k_deg(const int* __restrict__ dst,
                                             float* __restrict__ deg, int e) {
  int i = blockIdx.x * blockDim.x + threadIdx.x;
  if (i < e) atomicAdd(&deg[dst[i]], 1.0f);
}

__global__ __launch_bounds__(256) void k_rsqrt(float* __restrict__ p, int n) {
  int i = blockIdx.x * blockDim.x + threadIdx.x;
  if (i < n) p[i] = rsqrtf(p[i]);
}

// ---------------- WMMA GEMM: O[n x FOUT] = H[n x FIN] @ W[FIN x FOUT] ----------------
// 256-thread blocks = 8 wave32's, one 16x16 output tile per wave.
// FIN/FOUT compile-time: K-loop fully unrolled (FIN/32 WMMAs), static masks only.
// N is a multiple of 16 so no row guards anywhere; EXEC all-ones through WMMA.

template<int FIN, int FOUT>
__global__ __launch_bounds__(256) void k_gemm_wmma(const float* __restrict__ H,
                                                   const float* __restrict__ Wm,
                                                   float* __restrict__ O, int n) {
  const int lane = threadIdx.x & 31;
  const int wave = threadIdx.x >> 5;
  const int half = lane >> 4;
  const int l16  = lane & 15;
  const int row0 = (blockIdx.x * 8 + wave) * 16;
  if (row0 >= n) return;                       // uniform per wave
  const int n0 = blockIdx.y * 16;

  v8f c = {};
  const float* __restrict__ hrow = H + (long)(row0 + l16) * FIN;

#pragma unroll
  for (int k0 = 0; k0 < FIN; k0 += 32) {
    // ---- A fragment (16x32 f16): a[0..7] <- K=k0+8*half+{0..7}, a[8..15] <- K=k0+16+8*half+{0..7}
    v16h a;
#pragma unroll
    for (int g = 0; g < 2; ++g) {
      constexpr int GK = 16;
      if ((k0 + GK * 1 + 16 * g) <= FIN || (k0 + 16 * g + 16) <= FIN) {}  // (kept readable below)
      if ((k0 + 16 * g + 16) <= FIN) {          // compile-time: whole 16-K group in range
        const int kbase = k0 + 16 * g + 8 * half;
        const float4 v0 = *(const float4*)(hrow + kbase);
        const float4 v1 = *(const float4*)(hrow + kbase + 4);
        a[8*g+0] = (_Float16)v0.x; a[8*g+1] = (_Float16)v0.y;
        a[8*g+2] = (_Float16)v0.z; a[8*g+3] = (_Float16)v0.w;
        a[8*g+4] = (_Float16)v1.x; a[8*g+5] = (_Float16)v1.y;
        a[8*g+6] = (_Float16)v1.z; a[8*g+7] = (_Float16)v1.w;
      } else {
#pragma unroll
        for (int q = 0; q < 8; ++q) a[8*g+q] = (_Float16)0.0f;
      }
    }
    // ---- B fragment (32x16 f16): lane -> N = n0 + l16, element i -> K same as a[i]
    const int brow = k0 + 8 * half;
    const int bcol  = (FOUT >= 16 || l16 < FOUT) ? l16 : (FOUT - 1);
    const float cmask = (FOUT >= 16 || l16 < FOUT) ? 1.0f : 0.0f;
    v16h b;
#pragma unroll
    for (int i = 0; i < 16; ++i) {
      const int kk = brow + i + (i >= 8 ? 8 : 0);
      const int   brc   = (kk < FIN) ? kk : (FIN - 1);
      const float bmask = (kk < FIN) ? cmask : 0.0f;
      b[i] = (_Float16)(Wm[brc * FOUT + n0 + bcol] * bmask);
    }
    c = __builtin_amdgcn_wmma_f32_16x16x32_f16(false, a, false, b, (short)0, c, false, false);
  }

  // ---- D store: VGPR r -> row = row0 + r + 8*half, col = n0 + l16 (static col guard)
  if (FOUT >= 16 || l16 < FOUT) {
    float* __restrict__ op = O + (long)(row0 + 8 * half) * FOUT + n0 + l16;
#pragma unroll
    for (int r = 0; r < 8; ++r) op[r * FOUT] = c[r];
  }
}

// ---------------- aggregation ----------------

// agg = bias + dinv[i]^2 * hlin   (self-loop + bias, runs before the atomic scatter)
template<int F, int VEC>
__global__ __launch_bounds__(256) void k_agg_init(const float* __restrict__ hlin,
                                                  const float* __restrict__ dinv,
                                                  const float* __restrict__ bias,
                                                  float* __restrict__ agg, int n) {
  constexpr int CH = F / VEC;
  int idx = blockIdx.x * blockDim.x + threadIdx.x;
  if (idx >= n * CH) return;
  int i = idx / CH, c0 = (idx - i * CH) * VEC;
  float di = dinv[i];
  float d2 = di * di;
  const float* __restrict__ hp = hlin + (long)i * F + c0;
  float* __restrict__ ap = agg + (long)i * F + c0;
#pragma unroll
  for (int j = 0; j < VEC; ++j) ap[j] = bias[c0 + j] + hp[j] * d2;
}

// edge x feature-chunk parallel scatter-add; agg (<=25.6MB) is L2-resident
template<int F, int VEC>
__global__ __launch_bounds__(256) void k_scatter(const float* __restrict__ hlin,
                                                 const float* __restrict__ dinv,
                                                 const int* __restrict__ src,
                                                 const int* __restrict__ dst,
                                                 float* __restrict__ agg, int e) {
  constexpr int CH = F / VEC;
  int idx = blockIdx.x * blockDim.x + threadIdx.x;
  if (idx >= e * CH) return;
  int ed = idx / CH, c0 = (idx - ed * CH) * VEC;
  int s = src[ed], d = dst[ed];
  float w = dinv[s] * dinv[d];
  const float* __restrict__ hp = hlin + (long)s * F + c0;
  float* __restrict__ ap = agg + (long)d * F + c0;
  if constexpr (VEC == 4) {
    float4 v = *(const float4*)hp;
    atomicAdd(ap + 0, v.x * w);
    atomicAdd(ap + 1, v.y * w);
    atomicAdd(ap + 2, v.z * w);
    atomicAdd(ap + 3, v.w * w);
  } else {
    float2 v = *(const float2*)hp;
    atomicAdd(ap + 0, v.x * w);
    atomicAdd(ap + 1, v.y * w);
  }
}

// ---------------- BatchNorm + GELU ----------------

template<int F>
__global__ __launch_bounds__(256) void k_bn_partial(const float* __restrict__ agg,
                                                    float* __restrict__ sum,
                                                    float* __restrict__ sumsq, int n) {
  __shared__ float ls[64];
  __shared__ float lq[64];
  int tid = threadIdx.x;
  if (tid < F) { ls[tid] = 0.0f; lq[tid] = 0.0f; }
  __syncthreads();
  int base_row = blockIdx.x * BN_ROWS;
  int rows = n - base_row; if (rows > BN_ROWS) rows = BN_ROWS;
  int cnt = rows * F;
  const float* __restrict__ p = agg + (long)base_row * F;
  for (int idx = tid; idx < cnt; idx += 256) {
    float v = p[idx];
    int c = idx % F;                 // F is a power of two here -> AND
    atomicAdd(&ls[c], v);            // ds_add_f32
    atomicAdd(&lq[c], v * v);
  }
  __syncthreads();
  if (tid < F) { atomicAdd(&sum[tid], ls[tid]); atomicAdd(&sumsq[tid], lq[tid]); }
}

__global__ __launch_bounds__(64) void k_bn_finalize(const float* __restrict__ sum,
                                                    const float* __restrict__ sumsq,
                                                    const float* __restrict__ g,
                                                    const float* __restrict__ be,
                                                    float* __restrict__ scale,
                                                    float* __restrict__ shift,
                                                    int n, int f) {
  int t = threadIdx.x;
  if (t < f) {
    float inv_n = 1.0f / (float)n;
    float mu  = sum[t] * inv_n;
    float var = sumsq[t] * inv_n - mu * mu;
    float s   = g[t] * rsqrtf(var + 1e-5f);
    scale[t] = s;
    shift[t] = be[t] - mu * s;
  }
}

template<int F, int VEC>
__global__ __launch_bounds__(256) void k_bn_gelu(const float* __restrict__ agg,
                                                 const float* __restrict__ scale,
                                                 const float* __restrict__ shift,
                                                 float* __restrict__ out, int n) {
  constexpr int CH = F / VEC;
  int idx = blockIdx.x * blockDim.x + threadIdx.x;
  if (idx >= n * CH) return;
  int i = idx / CH, c0 = (idx - i * CH) * VEC;
  const float* __restrict__ ap = agg + (long)i * F + c0;
  float* __restrict__ op = out + (long)i * F + c0;
#pragma unroll
  for (int j = 0; j < VEC; ++j) {
    float y = ap[j] * scale[c0 + j] + shift[c0 + j];
    op[j] = 0.5f * y * (1.0f + erff(y * 0.70710678118654752f));  // exact GELU
  }
}

// ---------------- log_softmax over C=10 classes ----------------

__global__ __launch_bounds__(256) void k_logsoftmax(const float* __restrict__ agg,
                                                    float* __restrict__ out, int n) {
  constexpr int C = 10;
  int i = blockIdx.x * blockDim.x + threadIdx.x;
  if (i < n) {
    const float* r = agg + (long)i * C;
    float v[C];
#pragma unroll
    for (int j = 0; j < C; ++j) v[j] = r[j];
    float m = v[0];
#pragma unroll
    for (int j = 1; j < C; ++j) m = fmaxf(m, v[j]);
    float s = 0.0f;
#pragma unroll
    for (int j = 0; j < C; ++j) s += expf(v[j] - m);
    float l = m + logf(s);
    float* o = out + (long)i * C;
#pragma unroll
    for (int j = 0; j < C; ++j) o[j] = v[j] - l;
  }
}

// ---------------- host orchestration ----------------

template<int FIN, int FOUT, int VEC>
static void run_layer(const float* hin, const float* W, const float* b,
                      const float* g, const float* be, bool bn,
                      float* hlin, float* agg, float* sum, float* sumsq,
                      float* scale, float* shift, const float* dinv,
                      const int* srcp, const int* dstp, int N, int E,
                      hipStream_t stream) {
  dim3 gg((N / 16 + 7) / 8, (FOUT + 15) / 16);
  k_gemm_wmma<FIN, FOUT><<<gg, 256, 0, stream>>>(hin, W, hlin, N);

  constexpr int CH = FOUT / VEC;
  int nch = N * CH;
  k_agg_init<FOUT, VEC><<<(nch + 255) / 256, 256, 0, stream>>>(hlin, dinv, b, agg, N);
  long ech = (long)E * CH;
  k_scatter<FOUT, VEC><<<(int)((ech + 255) / 256), 256, 0, stream>>>(hlin, dinv, srcp, dstp, agg, E);

  if (bn) {
    k_fill<<<1, 256, 0, stream>>>(sum, 0.0f, 128);           // zero sum+sumsq (contiguous)
    int nb = (N + BN_ROWS - 1) / BN_ROWS;
    k_bn_partial<FOUT><<<nb, 256, 0, stream>>>(agg, sum, sumsq, N);
    k_bn_finalize<<<1, 64, 0, stream>>>(sum, sumsq, g, be, scale, shift, N, FOUT);
    k_bn_gelu<FOUT, VEC><<<(nch + 255) / 256, 256, 0, stream>>>(agg, scale, shift, hlin, N);
  }
}

extern "C" void kernel_launch(void* const* d_in, const int* in_sizes, int n_in,
                              void* d_out, int out_size, void* d_ws, size_t ws_size,
                              hipStream_t stream) {
  (void)in_sizes; (void)n_in; (void)out_size; (void)ws_size;
  const int N = 100000, E = 1600000;

  const float* x    = (const float*)d_in[0];
  const int*   ei   = (const int*)d_in[1];
  const int*   srcp = ei;        // edge_index[0]
  const int*   dstp = ei + E;    // edge_index[1]
  const float* W1 = (const float*)d_in[2],  *b1 = (const float*)d_in[3];
  const float* g1 = (const float*)d_in[4],  *be1 = (const float*)d_in[5];
  const float* W2 = (const float*)d_in[6],  *b2 = (const float*)d_in[7];
  const float* g2 = (const float*)d_in[8],  *be2 = (const float*)d_in[9];
  const float* W3 = (const float*)d_in[10], *b3 = (const float*)d_in[11];
  const float* g3 = (const float*)d_in[12], *be3 = (const float*)d_in[13];
  const float* W4 = (const float*)d_in[14], *b4 = (const float*)d_in[15];

  // workspace carve-out (256B aligned)
  char* w = (char*)d_ws;
  auto take = [&](size_t bytes) -> char* {
    char* p = w;
    w += (bytes + 255) & ~(size_t)255;
    return p;
  };
  float* dinv  = (float*)take(sizeof(float) * (size_t)N);
  float* stats = (float*)take(sizeof(float) * 256);
  float* bufA  = (float*)take(sizeof(float) * (size_t)N * 64);
  float* bufB  = (float*)take(sizeof(float) * (size_t)N * 64);
  float* sum = stats, *sumsq = stats + 64, *scale = stats + 128, *shift = stats + 192;

  // deg = 1 (self-loop) + in-degree; dinv = rsqrt(deg)
  k_fill <<<(N + 255) / 256, 256, 0, stream>>>(dinv, 1.0f, N);
  k_deg  <<<(E + 255) / 256, 256, 0, stream>>>(dstp, dinv, E);
  k_rsqrt<<<(N + 255) / 256, 256, 0, stream>>>(dinv, N);

  // buffer rotation: hin -> [gemm] hlin -> [scatter] agg -> [bn_gelu] back into hlin's buffer
  run_layer<128, 16, 4>(x,    W1, b1, g1, be1, true,  bufA, bufB, sum, sumsq, scale, shift,
                        dinv, srcp, dstp, N, E, stream);
  run_layer<16,  32, 4>(bufA, W2, b2, g2, be2, true,  bufB, bufA, sum, sumsq, scale, shift,
                        dinv, srcp, dstp, N, E, stream);
  run_layer<32,  64, 4>(bufB, W3, b3, g3, be3, true,  bufA, bufB, sum, sumsq, scale, shift,
                        dinv, srcp, dstp, N, E, stream);
  run_layer<64,  10, 2>(bufA, W4, b4, nullptr, nullptr, false, bufB, bufA, sum, sumsq, scale, shift,
                        dinv, srcp, dstp, N, E, stream);
  k_logsoftmax<<<(N + 255) / 256, 256, 0, stream>>>(bufA, (float*)d_out, N);
}
